// LiquidTimeConstantLayer_57475252355107
// MI455X (gfx1250) — matrix operand order repaired and checked by reading, not compiled, not run on hardware
//
#include <hip/hip_runtime.h>
#include <hip/hip_bf16.h>

typedef __bf16 bf16_t;
typedef __attribute__((ext_vector_type(16))) __bf16 v16bf;
typedef __attribute__((ext_vector_type(8)))  __bf16 v8bf;
typedef __attribute__((ext_vector_type(8)))  float  v8f;

#define DIM_I 512
#define DIM_H 1024
#define DIM_B 32
#define DIM_S 512
#define DIM_BS (DIM_B * DIM_S)   // 16384
#define DT_CONST 0.1f
#define RECUR_WGS 16

__device__ __forceinline__ float sigmoidf_fast(float x) {
    return 1.0f / (1.0f + __expf(-x));
}

// A fragment: 16x32 bf16 tile. Lane l<16 holds row M=l, K = {k0+0..7, k0+16..23};
// lane l>=16 holds row M=l-16, K = {k0+8..15, k0+24..31}.  (ISA 7.12.2)
__device__ __forceinline__ v16bf ld_frag_a(const bf16_t* __restrict__ row, int k0, int half) {
    v8bf lo = *(const v8bf*)(row + k0 + half * 8);
    v8bf hi = *(const v8bf*)(row + k0 + 16 + half * 8);
    return __builtin_shufflevector(lo, hi, 0,1,2,3,4,5,6,7,8,9,10,11,12,13,14,15);
}

// B fragment: 32x16 bf16 tile from transposed weights Wt (N x K row-major).
// Lane l holds column N=(l&15), K = {k0+half*8+0..7, k0+16+half*8+0..7} (same order as A).
__device__ __forceinline__ v16bf ld_frag_b(const bf16_t* __restrict__ wtrow, int k0, int half) {
    const bf16_t* p = wtrow + k0 + half * 8;
    v8bf lo = *(const v8bf*)p;
    v8bf hi = *(const v8bf*)(p + 16);
    return __builtin_shufflevector(lo, hi, 0,1,2,3,4,5,6,7,8,9,10,11,12,13,14,15);
}

// ---------------- conversion / init kernels ----------------

__global__ __launch_bounds__(256) void cvt_f32_bf16(const float* __restrict__ in,
                                                    bf16_t* __restrict__ out, int n) {
    int i = blockIdx.x * blockDim.x + threadIdx.x;
    if (i < n) out[i] = (bf16_t)in[i];
}

// in: K x N row-major f32 ; out: N x K row-major bf16 (transposed)
__global__ __launch_bounds__(256) void transpose_cvt(const float* __restrict__ in,
                                                     bf16_t* __restrict__ out, int K, int N) {
    int i = blockIdx.x * blockDim.x + threadIdx.x;
    if (i < N * K) {
        int n = i / K;
        int k = i - n * K;
        out[i] = (bf16_t)in[(size_t)k * N + n];
    }
}

__global__ __launch_bounds__(256) void init_h(const float* __restrict__ h_in,
                                              float* __restrict__ hf,
                                              bf16_t* __restrict__ hb,
                                              unsigned* __restrict__ barrier_ctr, int n) {
    int i = blockIdx.x * blockDim.x + threadIdx.x;
    if (i == 0) *barrier_ctr = 0u;
    if (i < n) {
        float v = h_in[i];
        hf[i] = v;
        hb[i] = (bf16_t)v;
    }
}

// ---------------- pre-phase: 4 fused GEMMs (16384x512)x(512x1024) ----------------
// One 16x16 output tile per wave; A fragment shared by 4 WMMA accumulators.

__global__ __launch_bounds__(256) void pre_gemm(
    const bf16_t* __restrict__ xb,
    const bf16_t* __restrict__ wt_sw, const bf16_t* __restrict__ wt_smu,
    const bf16_t* __restrict__ wt_ssig, const bf16_t* __restrict__ wt_tcx,
    const float* __restrict__ sw_b, const float* __restrict__ smu_b,
    const float* __restrict__ ssig_b, const float* __restrict__ tc_b,
    float* __restrict__ sens, float* __restrict__ tcx_out)
{
    const int wave  = blockIdx.x * (blockDim.x >> 5) + (threadIdx.x >> 5);
    const int lane  = threadIdx.x & 31;
    const int half  = lane >> 4;
    const int l16   = lane & 15;
    const int mtile = wave >> 6;   // 1024 tiles over M=16384
    const int ntile = wave & 63;   // 64 tiles over N=1024

    const bf16_t* arow = xb + (size_t)(mtile * 16 + l16) * DIM_I;
    const int nglob = ntile * 16 + l16;
    const bf16_t* b0 = wt_sw   + (size_t)nglob * DIM_I;
    const bf16_t* b1 = wt_smu  + (size_t)nglob * DIM_I;
    const bf16_t* b2 = wt_ssig + (size_t)nglob * DIM_I;
    const bf16_t* b3 = wt_tcx  + (size_t)nglob * DIM_I;

    v8f acc0 = {}, acc1 = {}, acc2 = {}, acc3 = {};
    for (int k0 = 0; k0 < DIM_I; k0 += 32) {
        v16bf a = ld_frag_a(arow, k0, half);
        v16bf bb;
        bb = ld_frag_b(b0, k0, half);
        acc0 = __builtin_amdgcn_wmma_f32_16x16x32_bf16(false, a, false, bb, (short)0, acc0, false, false);
        bb = ld_frag_b(b1, k0, half);
        acc1 = __builtin_amdgcn_wmma_f32_16x16x32_bf16(false, a, false, bb, (short)0, acc1, false, false);
        bb = ld_frag_b(b2, k0, half);
        acc2 = __builtin_amdgcn_wmma_f32_16x16x32_bf16(false, a, false, bb, (short)0, acc2, false, false);
        bb = ld_frag_b(b3, k0, half);
        acc3 = __builtin_amdgcn_wmma_f32_16x16x32_bf16(false, a, false, bb, (short)0, acc3, false, false);
    }

    const float bsw = sw_b[nglob], bsmu = smu_b[nglob], bssig = ssig_b[nglob], btc = tc_b[nglob];
    #pragma unroll
    for (int r = 0; r < 8; ++r) {
        const int m = mtile * 16 + half * 8 + r;          // D layout: M = r + 8*half
        const size_t o = (size_t)m * DIM_H + nglob;
        sens[o] = (acc0[r] + bsw) * sigmoidf_fast(acc1[r] + bsmu) * __expf(acc2[r] + bssig);
        tcx_out[o] = acc3[r] + btc;
    }
}

// ---------------- recurrent persistent kernel ----------------
// 16 WGs x 8 waves = 128 waves; wave = (mtile in 0..1, ntile in 0..63).
// Per step each wave computes 4 accumulators (tc, iw, imu, isig) over K=1024,
// applies the LTC update for its 16x16 tile of h, then device-wide barrier.

__global__ __launch_bounds__(256) void recur_scan(
    const bf16_t* __restrict__ wt_tch, const bf16_t* __restrict__ wt_iw,
    const bf16_t* __restrict__ wt_imu, const bf16_t* __restrict__ wt_isig,
    const float* __restrict__ iw_b, const float* __restrict__ imu_b,
    const float* __restrict__ isig_b,
    const float* __restrict__ sens, const float* __restrict__ tcx,
    float* __restrict__ h_f32, bf16_t* __restrict__ h_bf16,
    float* __restrict__ out, unsigned* __restrict__ barrier_ctr)
{
    const int wave  = blockIdx.x * (blockDim.x >> 5) + (threadIdx.x >> 5); // 0..127
    const int lane  = threadIdx.x & 31;
    const int half  = lane >> 4;
    const int l16   = lane & 15;
    const int mtile = wave >> 6;   // 0..1
    const int ntile = wave & 63;   // 0..63
    const int nglob = ntile * 16 + l16;

    const bf16_t* btc = wt_tch  + (size_t)nglob * DIM_H;
    const bf16_t* biw = wt_iw   + (size_t)nglob * DIM_H;
    const bf16_t* bmu = wt_imu  + (size_t)nglob * DIM_H;
    const bf16_t* bsg = wt_isig + (size_t)nglob * DIM_H;
    const float bb_iw = iw_b[nglob], bb_mu = imu_b[nglob], bb_sg = isig_b[nglob];

    for (int t = 0; t < DIM_S; ++t) {
        const int cur = t & 1, nxt = (t + 1) & 1;
        const bf16_t* arow = h_bf16 + (size_t)cur * DIM_B * DIM_H
                             + (size_t)(mtile * 16 + l16) * DIM_H;

        v8f a_tc = {}, a_iw = {}, a_mu = {}, a_sg = {};
        #pragma unroll 4
        for (int k0 = 0; k0 < DIM_H; k0 += 32) {
            v16bf a = ld_frag_a(arow, k0, half);
            v16bf w;
            w = ld_frag_b(btc, k0, half);
            a_tc = __builtin_amdgcn_wmma_f32_16x16x32_bf16(false, a, false, w, (short)0, a_tc, false, false);
            w = ld_frag_b(biw, k0, half);
            a_iw = __builtin_amdgcn_wmma_f32_16x16x32_bf16(false, a, false, w, (short)0, a_iw, false, false);
            w = ld_frag_b(bmu, k0, half);
            a_mu = __builtin_amdgcn_wmma_f32_16x16x32_bf16(false, a, false, w, (short)0, a_mu, false, false);
            w = ld_frag_b(bsg, k0, half);
            a_sg = __builtin_amdgcn_wmma_f32_16x16x32_bf16(false, a, false, w, (short)0, a_sg, false, false);
        }

        const float* hf_cur = h_f32 + (size_t)cur * DIM_B * DIM_H;
        float*       hf_nxt = h_f32 + (size_t)nxt * DIM_B * DIM_H;
        bf16_t*      hb_nxt = h_bf16 + (size_t)nxt * DIM_B * DIM_H;

        #pragma unroll
        for (int r = 0; r < 8; ++r) {
            const int b = mtile * 16 + half * 8 + r;                 // batch row
            const size_t eidx = (size_t)b * DIM_H + nglob;           // (b, n)
            const size_t sidx = ((size_t)b * DIM_S + t) * DIM_H + nglob; // (b, t, n)
            const float h_old = hf_cur[eidx];
            const float tau   = sigmoidf_fast(tcx[sidx] + a_tc[r]) + 0.1f;
            const float inter = (a_iw[r] + bb_iw) * sigmoidf_fast(a_mu[r] + bb_mu)
                                * __expf(a_sg[r] + bb_sg);
            const float h_new = h_old + DT_CONST * (sens[sidx] + inter - h_old) / tau;
            hf_nxt[eidx] = h_new;
            hb_nxt[eidx] = (bf16_t)h_new;
            out[sidx]    = h_new;
        }

        // device-wide barrier (monotonic counter; 16 WGs are always co-resident)
        __threadfence();
        __syncthreads();
        if (threadIdx.x == 0) {
            __hip_atomic_fetch_add(barrier_ctr, 1u, __ATOMIC_ACQ_REL, __HIP_MEMORY_SCOPE_AGENT);
            const unsigned target = (unsigned)(t + 1) * RECUR_WGS;
            while (__hip_atomic_load(barrier_ctr, __ATOMIC_ACQUIRE, __HIP_MEMORY_SCOPE_AGENT) < target)
                __builtin_amdgcn_s_sleep(2);
        }
        __syncthreads();
    }
}

// ---------------- host side ----------------

extern "C" void kernel_launch(void* const* d_in, const int* in_sizes, int n_in,
                              void* d_out, int out_size, void* d_ws, size_t ws_size,
                              hipStream_t stream) {
    const float* x      = (const float*)d_in[0];
    const float* h0     = (const float*)d_in[1];
    const float* sw_W   = (const float*)d_in[2];
    const float* sw_b   = (const float*)d_in[3];
    const float* smu_W  = (const float*)d_in[4];
    const float* smu_b  = (const float*)d_in[5];
    const float* ssig_W = (const float*)d_in[6];
    const float* ssig_b = (const float*)d_in[7];
    const float* iw_W   = (const float*)d_in[8];
    const float* iw_b   = (const float*)d_in[9];
    const float* imu_W  = (const float*)d_in[10];
    const float* imu_b  = (const float*)d_in[11];
    const float* isig_W = (const float*)d_in[12];
    const float* isig_b = (const float*)d_in[13];
    const float* tc_W   = (const float*)d_in[14];
    const float* tc_b   = (const float*)d_in[15];

    char* ws = (char*)d_ws;
    size_t off = 0;
    auto carve = [&](size_t bytes) { char* p = ws + off; off += (bytes + 255) & ~size_t(255); return p; };

    bf16_t* xb      = (bf16_t*)carve((size_t)DIM_BS * DIM_I * 2);
    bf16_t* wt_sw   = (bf16_t*)carve((size_t)DIM_H * DIM_I * 2);
    bf16_t* wt_smu  = (bf16_t*)carve((size_t)DIM_H * DIM_I * 2);
    bf16_t* wt_ssig = (bf16_t*)carve((size_t)DIM_H * DIM_I * 2);
    bf16_t* wt_tcx  = (bf16_t*)carve((size_t)DIM_H * DIM_I * 2);
    bf16_t* wt_tch  = (bf16_t*)carve((size_t)DIM_H * DIM_H * 2);
    bf16_t* wt_iw   = (bf16_t*)carve((size_t)DIM_H * DIM_H * 2);
    bf16_t* wt_imu  = (bf16_t*)carve((size_t)DIM_H * DIM_H * 2);
    bf16_t* wt_isig = (bf16_t*)carve((size_t)DIM_H * DIM_H * 2);
    float*  sens    = (float*)carve((size_t)DIM_BS * DIM_H * 4);
    float*  tcx     = (float*)carve((size_t)DIM_BS * DIM_H * 4);
    float*  h_f32   = (float*)carve((size_t)2 * DIM_B * DIM_H * 4);
    bf16_t* h_bf16  = (bf16_t*)carve((size_t)2 * DIM_B * DIM_H * 2);
    unsigned* bar   = (unsigned*)carve(256);

    // x -> bf16
    {
        int n = DIM_BS * DIM_I;
        cvt_f32_bf16<<<(n + 255) / 256, 256, 0, stream>>>(x, xb, n);
    }
    // transposed bf16 weights (N x K)
    {
        int n = DIM_H * DIM_I;
        int g = (n + 255) / 256;
        transpose_cvt<<<g, 256, 0, stream>>>(sw_W,   wt_sw,   DIM_I, DIM_H);
        transpose_cvt<<<g, 256, 0, stream>>>(smu_W,  wt_smu,  DIM_I, DIM_H);
        transpose_cvt<<<g, 256, 0, stream>>>(ssig_W, wt_ssig, DIM_I, DIM_H);
        transpose_cvt<<<g, 256, 0, stream>>>(tc_W,   wt_tcx,  DIM_I, DIM_H); // first I rows
    }
    {
        int n = DIM_H * DIM_H;
        int g = (n + 255) / 256;
        transpose_cvt<<<g, 256, 0, stream>>>(tc_W + (size_t)DIM_I * DIM_H, wt_tch, DIM_H, DIM_H);
        transpose_cvt<<<g, 256, 0, stream>>>(iw_W,   wt_iw,   DIM_H, DIM_H);
        transpose_cvt<<<g, 256, 0, stream>>>(imu_W,  wt_imu,  DIM_H, DIM_H);
        transpose_cvt<<<g, 256, 0, stream>>>(isig_W, wt_isig, DIM_H, DIM_H);
    }
    // h init + barrier reset
    {
        int n = DIM_B * DIM_H;
        init_h<<<(n + 255) / 256, 256, 0, stream>>>(h0, h_f32, h_bf16, bar, n);
    }
    // pre-phase: 65536 tiles, 8 waves/WG -> 8192 WGs
    pre_gemm<<<8192, 256, 0, stream>>>(xb, wt_sw, wt_smu, wt_ssig, wt_tcx,
                                       sw_b, smu_b, ssig_b, tc_b, sens, tcx);
    // recurrent scan: 16 persistent WGs
    recur_scan<<<RECUR_WGS, 256, 0, stream>>>(wt_tch, wt_iw, wt_imu, wt_isig,
                                              iw_b, imu_b, isig_b,
                                              sens, tcx, h_f32, h_bf16,
                                              (float*)d_out, bar);
    (void)in_sizes; (void)n_in; (void)out_size; (void)ws_size;
}
